// Qwen3TTSTalkerAttention_32006096290248
// MI455X (gfx1250) — matrix operand order repaired and checked by reading, not compiled, not run on hardware
//
#include <hip/hip_runtime.h>
#include <hip/hip_bf16.h>

// Problem constants (from reference)
#define BDIM 2
#define SEQ  2048
#define HIDN 2048
#define NH   16
#define NKV  8
#define DH   128
#define EPSV 1e-6f
#define SCALEV 0.08838834764831845f   // 128^-0.5

typedef __attribute__((ext_vector_type(16))) __bf16 v16bf;
typedef __attribute__((ext_vector_type(8)))  float  v8f;

// ---- Tensor Data Mover support (device pass only; host pass -> fallback) ----
#if defined(__has_builtin)
#if __has_builtin(__builtin_amdgcn_tensor_load_to_lds)
#define HAVE_TDM 1
#endif
#endif

#ifdef HAVE_TDM
typedef __attribute__((ext_vector_type(4))) unsigned int u32x4;
typedef __attribute__((ext_vector_type(8))) int          i32x8;
typedef __attribute__((ext_vector_type(4))) int          i32x4;

__device__ __forceinline__ unsigned lds_addr_of(const void* p) {
  // Flat LDS aperture: low 32 bits of the generic address are the LDS byte addr
  return (unsigned)(uintptr_t)p;
}

// Issue a 2D TDM tile load: tile_d1 rows x tile_d0 elements (2-byte elements),
// row stride stride0 elements, from global gptr into LDS at lds_addr.
__device__ __forceinline__ void tdm_load_tile_2d(unsigned lds_addr, const void* gptr,
                                                 unsigned tile_d0, unsigned tile_d1,
                                                 unsigned tensor_d0, unsigned tensor_d1,
                                                 unsigned stride0) {
  const unsigned long long ga = (unsigned long long)(uintptr_t)gptr;
  u32x4 g0;
  g0[0] = 1u;                                              // count=1, user mode
  g0[1] = lds_addr;                                        // lds_addr (bytes)
  g0[2] = (unsigned)ga;                                    // global_addr[31:0]
  g0[3] = (unsigned)((ga >> 32) & 0x01ffffffu) | (2u << 30); // [56:32] | type=2
  i32x8 g1;
  g1[0] = (int)(1u << 16);                                 // data_size=1 (2B/elem)
  g1[1] = (int)((tensor_d0 & 0xffffu) << 16);              // tensor_dim0[15:0]
  g1[2] = (int)((tensor_d0 >> 16) | ((tensor_d1 & 0xffffu) << 16));
  g1[3] = (int)((tensor_d1 >> 16) | (tile_d0 << 16));      // tile_dim0
  g1[4] = (int)(tile_d1 & 0xffffu);                        // tile_dim1 (dim2=0)
  g1[5] = (int)stride0;                                    // dim0_stride[31:0]
  g1[6] = 0;
  g1[7] = 0;
  i32x4 z4; z4[0] = 0; z4[1] = 0; z4[2] = 0; z4[3] = 0;
#if __clang_major__ >= 23
  i32x8 z8;
  z8[0] = 0; z8[1] = 0; z8[2] = 0; z8[3] = 0;
  z8[4] = 0; z8[5] = 0; z8[6] = 0; z8[7] = 0;
  __builtin_amdgcn_tensor_load_to_lds(g0, g1, z4, z4, z8, 0);
#else
  __builtin_amdgcn_tensor_load_to_lds(g0, g1, z4, z4, 0);
#endif
}
#endif  // HAVE_TDM

__device__ __forceinline__ v8f wmma_bf16f32(v16bf a, v16bf b, v8f c) {
  // D = A(16x32 bf16) * B(32x16 bf16) + C(16x16 f32)
  return __builtin_amdgcn_wmma_f32_16x16x32_bf16(false, a, false, b, (short)0, c,
                                                 false, false);
}

// A-matrix (16x32 bf16) per-lane fragment: lane L holds row (L&15); K values are
// kb..kb+15 with kb = (L>>4)*16.  Caller passes p = &row[kb].
__device__ __forceinline__ v16bf load_a_frag(const __bf16* __restrict__ p) {
  v16bf r;
#pragma unroll
  for (int i = 0; i < 16; ++i) r[i] = p[i];
  return r;
}

// B-matrix (32x16 bf16) per-lane fragment: lane L holds col (L&15); K values are
// kb..kb+15 with kb = (L>>4)*16.  Caller passes p = &col_data[kb] (contiguous).
__device__ __forceinline__ v16bf load_b_frag(const __bf16* __restrict__ p) {
  v16bf r;
#pragma unroll
  for (int i = 0; i < 16; ++i) r[i] = p[i];
  return r;
}

// -------------------------------------------------------------------------
// C[M,N] (f32) = A[M,K] * W[N,K]^T     (A fp32 or bf16, W fp32; bf16 WMMA math)
// block = 256 threads (8 waves); tile BM=128, BN=64, BK=32; wave -> 32x32 patch
// -------------------------------------------------------------------------
template <bool A_BF16>
__global__ __launch_bounds__(256) void gemm_bf16wmma(const void* __restrict__ Ap,
                                                     const float* __restrict__ W,
                                                     float* __restrict__ C,
                                                     int M, int N, int K) {
  constexpr int BM = 128, BN = 64, BK = 32;
  __shared__ __bf16 As[BM * BK];
  __shared__ __bf16 Bs[BN * BK];

  const int tid  = threadIdx.x;
  const int lane = tid & 31;
  const int wave = tid >> 5;
  const int m0 = blockIdx.y * BM;
  const int n0 = blockIdx.x * BN;
  const int wm = (wave >> 1) * 32;
  const int wn = (wave & 1) * 32;
  const int col   = lane & 15;
  const int kb8   = (lane >> 4) * 8;
  const int kb16  = (lane >> 4) * 16;
  const int rbase = (lane >> 4) * 8;

  v8f acc[2][2];
#pragma unroll
  for (int i = 0; i < 2; ++i)
#pragma unroll
    for (int j = 0; j < 2; ++j)
#pragma unroll
      for (int v = 0; v < 8; ++v) acc[i][j][v] = 0.0f;

  for (int k0 = 0; k0 < K; k0 += BK) {
    {  // stage A tile (convert fp32 -> bf16 on the fly if needed)
      const int row = tid >> 1;          // 0..127
      const int cb  = (tid & 1) * 16;
      if (A_BF16) {
        const __bf16* src = (const __bf16*)Ap + (size_t)(m0 + row) * K + k0 + cb;
#pragma unroll
        for (int i = 0; i < 16; ++i) As[row * BK + cb + i] = src[i];
        if (k0 + BK < K) __builtin_prefetch(src + BK, 0, 1);
      } else {
        const float* src = (const float*)Ap + (size_t)(m0 + row) * K + k0 + cb;
#pragma unroll
        for (int i = 0; i < 16; ++i) As[row * BK + cb + i] = (__bf16)src[i];
        if (k0 + BK < K) __builtin_prefetch(src + BK, 0, 1);
      }
    }
    {  // stage W tile (row n of W = column n of B)
      const int row = tid >> 2;          // 0..63
      const int cb  = (tid & 3) * 8;
      const float* src = W + (size_t)(n0 + row) * K + k0 + cb;
#pragma unroll
      for (int i = 0; i < 8; ++i) Bs[row * BK + cb + i] = (__bf16)src[i];
      if (k0 + BK < K) __builtin_prefetch(src + BK, 0, 1);
    }
    __syncthreads();

    v16bf af[2], bf[2];
#pragma unroll
    for (int i = 0; i < 2; ++i)
      af[i] = load_a_frag(&As[(wm + i * 16 + col) * BK + kb16]);
#pragma unroll
    for (int j = 0; j < 2; ++j)
      bf[j] = load_b_frag(&Bs[(wn + j * 16 + col) * BK + kb16]);
#pragma unroll
    for (int i = 0; i < 2; ++i)
#pragma unroll
      for (int j = 0; j < 2; ++j)
        acc[i][j] = wmma_bf16f32(af[i], bf[j], acc[i][j]);
    __syncthreads();
  }

  // C/D layout: lanes 0-15 -> rows v, lanes 16-31 -> rows v+8; col = lane&15
#pragma unroll
  for (int i = 0; i < 2; ++i)
#pragma unroll
    for (int j = 0; j < 2; ++j)
#pragma unroll
      for (int v = 0; v < 8; ++v) {
        const int row = m0 + wm + i * 16 + v + rbase;
        const int cc  = n0 + wn + j * 16 + col;
        C[(size_t)row * N + cc] = acc[i][j][v];
      }
}

// -------------------------------------------------------------------------
// RMSNorm + 3D mRoPE + relayout [B,S,nh,D]f32 -> [B,nh,S,D]bf16.
// One wave per (token, head); lane owns 4 consecutive d values.
// rotate_half partner d <-> d^64 lives in lane <-> lane^16.
// -------------------------------------------------------------------------
__global__ __launch_bounds__(256) void postproc_rope(const float* __restrict__ src,
                                                     const float* __restrict__ cosb,
                                                     const float* __restrict__ sinb,
                                                     const float* __restrict__ normw,
                                                     __bf16* __restrict__ dst,
                                                     int nheads, int do_rope) {
  const int gw   = (blockIdx.x * 256 + threadIdx.x) >> 5;
  const int lane = threadIdx.x & 31;
  const int h  = gw % nheads;
  const int bs = gw / nheads;
  const int b  = bs / SEQ;
  const int s  = bs % SEQ;
  const int dbase = lane * 4;

  const float* x = src + ((size_t)bs * nheads + h) * DH + dbase;
  float v[4];
#pragma unroll
  for (int i = 0; i < 4; ++i) v[i] = x[i];

  __bf16* out = dst + (((size_t)(b * nheads + h) * SEQ) + s) * DH + dbase;

  if (!do_rope) {  // V path: convert + relayout only
#pragma unroll
    for (int i = 0; i < 4; ++i) out[i] = (__bf16)v[i];
    return;
  }

  float ss = v[0] * v[0] + v[1] * v[1] + v[2] * v[2] + v[3] * v[3];
#pragma unroll
  for (int off = 16; off >= 1; off >>= 1) ss += __shfl_xor(ss, off, 32);
  const float r = rsqrtf(ss * (1.0f / DH) + EPSV);

  float xn[4];
#pragma unroll
  for (int i = 0; i < 4; ++i) xn[i] = v[i] * r * normw[dbase + i];

  float o4[4];
#pragma unroll
  for (int i = 0; i < 4; ++i) {
    const int d = dbase + i;
    const float p  = __shfl_xor(xn[i], 16, 32);
    const float rh = (lane < 16) ? -p : p;          // rotate_half
    const int dd = d & 63;                          // sections [21,21,22] repeat
    const int t  = (dd < 21) ? 0 : ((dd < 42) ? 1 : 2);
    const size_t ci = (((size_t)t * BDIM + b) * SEQ + s) * DH + d;
    o4[i] = xn[i] * cosb[ci] + rh * sinb[ci];
  }
#pragma unroll
  for (int i = 0; i < 4; ++i) out[i] = (__bf16)o4[i];
}

// -------------------------------------------------------------------------
// Causal flash attention (GQA). Block = 128 queries of one (b,h); 8 waves,
// each wave owns 16 query rows. Key loop in 32-key tiles.
//   K tile: TDM (tensor_load_to_lds) double-buffered, issued by wave 0,
//           synced with s_wait_tensorcnt (VALU-staged fallback if no TDM).
//   V tile: staged transposed in LDS so PV B-fragments are contiguous reads.
// Online softmax in fp32; QK^T and PV via bf16 WMMA.
// -------------------------------------------------------------------------
__global__ __launch_bounds__(256) void flash_attn(const __bf16* __restrict__ Q,
                                                  const __bf16* __restrict__ Kg,
                                                  const __bf16* __restrict__ Vg,
                                                  __bf16* __restrict__ O) {
  __shared__ __bf16 Kt[2][32 * DH];   // [buf][key][d]  (double buffer for TDM)
  __shared__ __bf16 VT[DH * 32];      // [d][key]
  __shared__ __bf16 Pbuf[8][16 * 32]; // per-wave P relayout buffer

  const int tid  = threadIdx.x;
  const int lane = tid & 31;
  const int wave = tid >> 5;
  const int bh   = blockIdx.y;
  const int b    = bh / NH;
  const int h    = bh % NH;
  const int kvh  = h / (NH / NKV);
  const int qblk = blockIdx.x;
  const int q0   = qblk * 128 + wave * 16;
  const int col   = lane & 15;
  const int kb8   = (lane >> 4) * 8;
  const int kb16  = (lane >> 4) * 16;
  const int rbase = (lane >> 4) * 8;

  const __bf16* Qm = Q  + ((size_t)(b * NH  + h)   * SEQ) * DH;
  const __bf16* Km = Kg + ((size_t)(b * NKV + kvh) * SEQ) * DH;
  const __bf16* Vm = Vg + ((size_t)(b * NKV + kvh) * SEQ) * DH;

  // Q fragments for all 4 K-steps of D=128, loaded once
  v16bf qf[4];
#pragma unroll
  for (int kd = 0; kd < 4; ++kd)
    qf[kd] = load_a_frag(Qm + (size_t)(q0 + col) * DH + kd * 32 + kb16);

  float mstate[8], lstate[8];
  v8f oacc[8];
#pragma unroll
  for (int v = 0; v < 8; ++v) { mstate[v] = -3.0e38f; lstate[v] = 0.0f; }
#pragma unroll
  for (int t = 0; t < 8; ++t)
#pragma unroll
    for (int v = 0; v < 8; ++v) oacc[t][v] = 0.0f;

  const int kend = qblk * 128 + 128;   // causal upper bound for this block

#ifdef HAVE_TDM
  if (wave == 0)   // prologue: DMA first K tile into buffer 0
    tdm_load_tile_2d(lds_addr_of(&Kt[0][0]), Km, DH, 32, DH, 32, DH);
#endif

  int buf = 0;
  for (int key0 = 0; key0 < kend; key0 += 32, buf ^= 1) {
    {  // cooperative V staging (transposed); K fallback staging if no TDM
      const int r  = tid >> 3;          // 0..31 key row
      const int cb = (tid & 7) * 16;    // d chunk
      const __bf16* vsrc = Vm + (size_t)(key0 + r) * DH + cb;
#pragma unroll
      for (int i = 0; i < 16; ++i) VT[(cb + i) * 32 + r] = vsrc[i];
      if (key0 + 32 < kend) __builtin_prefetch(vsrc + 32 * DH, 0, 1);
#ifndef HAVE_TDM
      const __bf16* ksrc = Km + (size_t)(key0 + r) * DH + cb;
#pragma unroll
      for (int i = 0; i < 16; ++i) Kt[0][r * DH + cb + i] = ksrc[i];
#endif
    }
#ifdef HAVE_TDM
    if (wave == 0) __builtin_amdgcn_s_wait_tensorcnt(0);  // K tile landed
#endif
    __syncthreads();
#ifdef HAVE_TDM
    if (wave == 0 && key0 + 32 < kend)   // DMA next K tile into other buffer
      tdm_load_tile_2d(lds_addr_of(&Kt[buf ^ 1][0]),
                       Km + (size_t)(key0 + 32) * DH, DH, 32, DH, 32, DH);
    const __bf16* KtC = &Kt[buf][0];
#else
    const __bf16* KtC = &Kt[0][0];
#endif

    // S = Q K^T  (16 queries x 32 keys), fp32 accum
    v8f sc[2];
#pragma unroll
    for (int j = 0; j < 2; ++j)
#pragma unroll
      for (int v = 0; v < 8; ++v) sc[j][v] = 0.0f;
#pragma unroll
    for (int kd = 0; kd < 4; ++kd)
#pragma unroll
      for (int j = 0; j < 2; ++j) {
        v16bf bf = load_b_frag(&KtC[(j * 16 + col) * DH + kd * 32 + kb16]);
        sc[j] = wmma_bf16f32(qf[kd], bf, sc[j]);
      }

    // online softmax (row = v + rbase within wave's 16 queries)
    float p0a[8], p1a[8], corr[8];
#pragma unroll
    for (int v = 0; v < 8; ++v) {
      const int qg = q0 + v + rbase;
      float s0 = sc[0][v] * SCALEV;
      float s1 = sc[1][v] * SCALEV;
      if (key0 + col > qg)      s0 = -3.0e38f;    // causal mask
      if (key0 + 16 + col > qg) s1 = -3.0e38f;
      float rm = fmaxf(s0, s1);
#pragma unroll
      for (int off = 8; off >= 1; off >>= 1)      // stays within 16-lane half
        rm = fmaxf(rm, __shfl_xor(rm, off, 32));
      const float mnew = fmaxf(mstate[v], rm);
      corr[v] = __expf(mstate[v] - mnew);
      p0a[v]  = __expf(s0 - mnew);
      p1a[v]  = __expf(s1 - mnew);
      float rs = p0a[v] + p1a[v];
#pragma unroll
      for (int off = 8; off >= 1; off >>= 1) rs += __shfl_xor(rs, off, 32);
      lstate[v] = lstate[v] * corr[v] + rs;
      mstate[v] = mnew;
    }
#pragma unroll
    for (int t = 0; t < 8; ++t)
#pragma unroll
      for (int v = 0; v < 8; ++v) oacc[t][v] *= corr[v];

    // C-layout -> A-layout relayout of P through per-wave LDS, as bf16
    __bf16* pb = &Pbuf[wave][0];
#pragma unroll
    for (int v = 0; v < 8; ++v) {
      pb[(v + rbase) * 32 + col]      = (__bf16)p0a[v];
      pb[(v + rbase) * 32 + 16 + col] = (__bf16)p1a[v];
    }
    v16bf pf = load_a_frag(&Pbuf[wave][col * 32 + kb16]);

    // O += P V  (8 d-tiles of 16)
#pragma unroll
    for (int t = 0; t < 8; ++t) {
      v16bf bf = load_b_frag(&VT[(t * 16 + col) * 32 + kb16]);
      oacc[t] = wmma_bf16f32(pf, bf, oacc[t]);
    }
    __syncthreads();
  }

  // normalize and store bf16 into [B,S,H*D]
#pragma unroll
  for (int t = 0; t < 8; ++t)
#pragma unroll
    for (int v = 0; v < 8; ++v) {
      const float o = oacc[t][v] / lstate[v];
      const int srow = q0 + v + rbase;
      O[((size_t)b * SEQ + srow) * (NH * DH) + h * DH + t * 16 + col] = (__bf16)o;
    }
}

// -------------------------------------------------------------------------
extern "C" void kernel_launch(void* const* d_in, const int* in_sizes, int n_in,
                              void* d_out, int out_size, void* d_ws, size_t ws_size,
                              hipStream_t stream) {
  (void)in_sizes; (void)n_in; (void)out_size; (void)ws_size;
  const float* hidden = (const float*)d_in[0];
  const float* cosb   = (const float*)d_in[1];
  const float* sinb   = (const float*)d_in[2];
  const float* q_w    = (const float*)d_in[3];
  const float* k_w    = (const float*)d_in[4];
  const float* v_w    = (const float*)d_in[5];
  const float* o_w    = (const float*)d_in[6];
  const float* q_nw   = (const float*)d_in[7];
  const float* k_nw   = (const float*)d_in[8];
  float* out = (float*)d_out;

  const int M = BDIM * SEQ;   // 4096 tokens
  char* ws = (char*)d_ws;
  // workspace layout (~112 MB)
  float*  qf32  = (float*)(ws + 0);                    // M x 2048 f32
  float*  kf32  = (float*)(ws + 33554432u);            // M x 1024 f32
  float*  vf32  = (float*)(ws + 50331648u);            // M x 1024 f32
  __bf16* qb    = (__bf16*)(ws + 67108864u);           // [B,H, S,D] bf16
  __bf16* kb    = (__bf16*)(ws + 83886080u);           // [B,KV,S,D] bf16
  __bf16* vb    = (__bf16*)(ws + 92274688u);           // [B,KV,S,D] bf16
  __bf16* attnb = (__bf16*)(ws + 100663296u);          // [B,S,H*D] bf16

  dim3 blk(256);

  // QKV projections (bf16 WMMA GEMMs, fp32 accumulate)
  gemm_bf16wmma<false><<<dim3(2048 / 64, M / 128), blk, 0, stream>>>(hidden, q_w, qf32, M, 2048, HIDN);
  gemm_bf16wmma<false><<<dim3(1024 / 64, M / 128), blk, 0, stream>>>(hidden, k_w, kf32, M, 1024, HIDN);
  gemm_bf16wmma<false><<<dim3(1024 / 64, M / 128), blk, 0, stream>>>(hidden, v_w, vf32, M, 1024, HIDN);

  // RMSNorm + 3D mRoPE + relayout (Q, K); convert/relayout (V)
  postproc_rope<<<dim3((M * NH)  / 8), blk, 0, stream>>>(qf32, cosb, sinb, q_nw, qb, NH,  1);
  postproc_rope<<<dim3((M * NKV) / 8), blk, 0, stream>>>(kf32, cosb, sinb, k_nw, kb, NKV, 1);
  postproc_rope<<<dim3((M * NKV) / 8), blk, 0, stream>>>(vf32, cosb, sinb, k_nw, vb, NKV, 0);

  // causal GQA flash attention (WMMA + TDM K-tile pipeline)
  flash_attn<<<dim3(SEQ / 128, BDIM * NH), blk, 0, stream>>>(qb, kb, vb, attnb);

  // output projection -> fp32 d_out
  gemm_bf16wmma<true><<<dim3(HIDN / 64, M / 128), blk, 0, stream>>>(attnb, o_w, out, M, HIDN, NH * DH);
}
